// LiquidS4Cell_4973572129222
// MI455X (gfx1250) — matrix operand order repaired and checked by reading, not compiled, not run on hardware
//
#include <hip/hip_runtime.h>
#include <hip/hip_bf16.h>

// ---------------------------------------------------------------------------
// LiquidS4 cell for MI455X (gfx1250):
//   GEMM1 (bf16 WMMA, B staged in LDS, double-buffered) : Bu = u @ B^T
//   chunked scan: x_t = a*x_{t-1} + Bu_t,  a = exp(-dt*exp(Lambda))
//   GEMM2 (bf16 WMMA, C staged in LDS): y = xs @ C^T + D*u
// ---------------------------------------------------------------------------

typedef __attribute__((ext_vector_type(16))) __bf16 v16bf;
typedef __attribute__((ext_vector_type(8)))  __bf16 v8bf;
typedef __attribute__((ext_vector_type(8)))  float  v8f;

static constexpr int BATCH  = 4;
static constexpr int SEQ    = 4096;
static constexpr int DM     = 1024;           // d_model
static constexpr int NS     = 64;             // state size N
static constexpr int M      = BATCH * SEQ;    // 16384 flattened rows
static constexpr int CHUNK  = 64;             // scan chunk length
static constexpr int NCHUNK = SEQ / CHUNK;    // 64 chunks per batch

static constexpr int SLDS   = 72;             // padded LDS row stride (bf16)

// ------------------------- weight conversion f32 -> bf16 -------------------
__global__ void convert_weights(const float* __restrict__ B,
                                const float* __restrict__ C,
                                __bf16* __restrict__ Bb,
                                __bf16* __restrict__ Cb) {
  int i = blockIdx.x * blockDim.x + threadIdx.x;     // grid covers 2*NS*DM
  const int total = NS * DM;                         // 65536
  if (i < total)            Bb[i]         = (__bf16)B[i];
  else                      Cb[i - total] = (__bf16)C[i - total];
}

// ------------------------- GEMM1: Bu = u @ B^T  (M x 64, K = 1024) ---------
// Block = 128 threads = 4 waves; wave tile = 32 rows x 64 cols (8 WMMA accs).
// B staged through double-buffered LDS K-slabs of 64.
__global__ void __launch_bounds__(128)
gemm1_bu(const float* __restrict__ u, const __bf16* __restrict__ Bb,
         float* __restrict__ Bu) {
  __shared__ __bf16 sh[2][NS * SLDS];         // 2 x (64 n-rows x 64 k), 18 KB

  const int tid  = threadIdx.x;
  const int wave = tid >> 5;
  const int lane = tid & 31;
  const int l16  = lane & 15;
  const int hi   = lane >> 4;                 // half-wave select
  const int rbase = blockIdx.x * 128 + wave * 32;

  const float* ArowLo = u + (size_t)(rbase + l16) * DM;        // rows 0..15
  const float* ArowHi = ArowLo + (size_t)16 * DM;              // rows 16..31

  // slab prefetch mapping: thread i -> B row n = i>>1, k chunk (i&1)*32 (64 B)
  const int pn = tid >> 1;
  const int pk = (tid & 1) * 32;
  const __bf16* gB = Bb + (size_t)pn * DM + pk;

  v8f aL0 = {}, aL1 = {}, aL2 = {}, aL3 = {};
  v8f aH0 = {}, aH1 = {}, aH2 = {}, aH3 = {};

  // preload slab 0
  v16bf p0 = *(const v16bf*)(gB);
  v16bf p1 = *(const v16bf*)(gB + 16);
  *(v16bf*)&sh[0][pn * SLDS + pk]      = p0;
  *(v16bf*)&sh[0][pn * SLDS + pk + 16] = p1;
  __syncthreads();

  for (int ks = 0; ks < DM / 64; ks++) {
    const int cur = ks & 1;
    // prefetch next slab into registers (overlaps with WMMA below)
    if (ks < DM / 64 - 1) {
      p0 = *(const v16bf*)(gB + (ks + 1) * 64);
      p1 = *(const v16bf*)(gB + (ks + 1) * 64 + 16);
    }
#pragma unroll
    for (int sub = 0; sub < 2; sub++) {
      const int kb = ks * 64 + sub * 32;       // global k base
      const int lk = sub * 32;                 // k base within slab
      // A fragments (two 16x32 row-groups), f32 loads + in-register bf16 cvt
      v8f f0 = *(const v8f*)(ArowLo + kb + hi * 16);
      v8f f1 = *(const v8f*)(ArowLo + kb + hi * 16 + 8);
      v8f f2 = *(const v8f*)(ArowHi + kb + hi * 16);
      v8f f3 = *(const v8f*)(ArowHi + kb + hi * 16 + 8);
      v16bf aLo, aHi;
#pragma unroll
      for (int i = 0; i < 8; i++) {
        aLo[i] = (__bf16)f0[i]; aLo[8 + i] = (__bf16)f1[i];
        aHi[i] = (__bf16)f2[i]; aHi[8 + i] = (__bf16)f3[i];
      }
      // B fragments from LDS: lane = column n, 16 contiguous K (32 B)
      const __bf16* sp = &sh[cur][l16 * SLDS + lk + hi * 16];
      v16bf b0 = *(const v16bf*)(sp);
      v16bf b1 = *(const v16bf*)(sp + 16 * SLDS);
      v16bf b2 = *(const v16bf*)(sp + 32 * SLDS);
      v16bf b3 = *(const v16bf*)(sp + 48 * SLDS);

      aL0 = __builtin_amdgcn_wmma_f32_16x16x32_bf16(false, aLo, false, b0, (short)0, aL0, false, false);
      aH0 = __builtin_amdgcn_wmma_f32_16x16x32_bf16(false, aHi, false, b0, (short)0, aH0, false, false);
      aL1 = __builtin_amdgcn_wmma_f32_16x16x32_bf16(false, aLo, false, b1, (short)0, aL1, false, false);
      aH1 = __builtin_amdgcn_wmma_f32_16x16x32_bf16(false, aHi, false, b1, (short)0, aH1, false, false);
      aL2 = __builtin_amdgcn_wmma_f32_16x16x32_bf16(false, aLo, false, b2, (short)0, aL2, false, false);
      aH2 = __builtin_amdgcn_wmma_f32_16x16x32_bf16(false, aHi, false, b2, (short)0, aH2, false, false);
      aL3 = __builtin_amdgcn_wmma_f32_16x16x32_bf16(false, aLo, false, b3, (short)0, aL3, false, false);
      aH3 = __builtin_amdgcn_wmma_f32_16x16x32_bf16(false, aHi, false, b3, (short)0, aH3, false, false);
    }
    // write next slab to the other buffer; one barrier per slab is sufficient
    if (ks < DM / 64 - 1) {
      const int nxt = cur ^ 1;
      *(v16bf*)&sh[nxt][pn * SLDS + pk]      = p0;
      *(v16bf*)&sh[nxt][pn * SLDS + pk + 16] = p1;
      __syncthreads();
    }
  }

  // C/D layout: VGPR r <-> (row = r + 8*hi, col = l16) in each 16x16 tile
#pragma unroll
  for (int r = 0; r < 8; r++) {
    const int rowL = rbase + r + hi * 8;
    const int rowH = rowL + 16;
    float* oL = Bu + (size_t)rowL * NS + l16;
    float* oH = Bu + (size_t)rowH * NS + l16;
    oL[0] = aL0[r]; oL[16] = aL1[r]; oL[32] = aL2[r]; oL[48] = aL3[r];
    oH[0] = aH0[r]; oH[16] = aH1[r]; oH[32] = aH2[r]; oH[48] = aH3[r];
  }
}

// ------------------------- scan phase 1: local chunk scans -----------------
__global__ void scan_local(float* __restrict__ Bu, float* __restrict__ S,
                           const float* __restrict__ Lambda,
                           const float* __restrict__ log_dt) {
  const int n  = threadIdx.x;
  const int bc = blockIdx.x;                       // b*NCHUNK + c
  const float dt = __expf(log_dt[0]);
  const float a  = __expf(-dt * __expf(Lambda[n]));
  const size_t base = (size_t)bc * CHUNK * NS + n;
  float x = 0.f;
#pragma unroll 4
  for (int t = 0; t < CHUNK; t++) {
    x = a * x + Bu[base + (size_t)t * NS];
    Bu[base + (size_t)t * NS] = x;
  }
  S[(size_t)bc * NS + n] = x;                      // chunk carry (x0 = 0)
}

// ------------------------- scan phase 2: carry scan over chunks ------------
__global__ void scan_carries(const float* __restrict__ S, float* __restrict__ Cin,
                             const float* __restrict__ Lambda,
                             const float* __restrict__ log_dt) {
  const int n = threadIdx.x;
  const int b = blockIdx.x;
  const float dt  = __expf(log_dt[0]);
  const float a64 = __expf(-(float)CHUNK * dt * __expf(Lambda[n]));  // a^CHUNK
  float acc = 0.f;
  for (int c = 0; c < NCHUNK; c++) {
    const size_t idx = ((size_t)b * NCHUNK + c) * NS + n;
    Cin[idx] = acc;
    acc = a64 * acc + S[idx];
  }
}

// ------------------------- scan phase 3: fixup + bf16 emit -----------------
__global__ void fixup_bf16(const float* __restrict__ Bu, const float* __restrict__ Cin,
                           const float* __restrict__ Lambda,
                           const float* __restrict__ log_dt,
                           __bf16* __restrict__ Xb) {
  const int idx = blockIdx.x * blockDim.x + threadIdx.x;   // < M*NS
  const int n  = idx & (NS - 1);
  const int m  = idx >> 6;
  const int tl = m & (CHUNK - 1);
  const int bc = m >> 6;                                   // b*NCHUNK + c
  const float dt = __expf(log_dt[0]);
  const float pw = __expf(-(float)(tl + 1) * dt * __expf(Lambda[n])); // a^(tl+1)
  Xb[idx] = (__bf16)(Bu[idx] + pw * Cin[(size_t)bc * NS + n]);
}

// ------------------------- GEMM2: y = xs @ C^T + D*u  (M x 1024, K = 64) ---
// Block = 256 threads = 8 waves; wave tile = 16 rows x 64 d-cols.
// The block's whole C-operand tile (64 d-rows x 64 n = 8 KB) staged in LDS once.
__global__ void __launch_bounds__(256)
gemm2_y(const __bf16* __restrict__ Xb, const __bf16* __restrict__ Cb,
        const float* __restrict__ u, const float* __restrict__ Dv,
        float* __restrict__ y) {
  __shared__ __bf16 shc[64 * SLDS];           // 64 d-rows x 64 n, padded, 9 KB

  const int tid  = threadIdx.x;
  const int wave = tid >> 5;
  const int lane = tid & 31;
  const int l16  = lane & 15;
  const int hi   = lane >> 4;
  const int rbase = blockIdx.x * 128 + wave * 16;
  const int dbase = blockIdx.y * 64;

  // stage C tile: thread i -> d-row i>>2, 16-elem chunk (i&3)*16 (32 B each)
  {
    const int cr = tid >> 2;
    const int ck = (tid & 3) * 16;
    *(v16bf*)&shc[cr * SLDS + ck] =
        *(const v16bf*)(Cb + (size_t)(dbase + cr) * NS + ck);
  }
  __syncthreads();

  const __bf16* Arow = Xb + (size_t)(rbase + l16) * NS;
  v8f acc0 = {}, acc1 = {}, acc2 = {}, acc3 = {};

#pragma unroll
  for (int kb = 0; kb < NS; kb += 32) {
    v8bf lo = *(const v8bf*)(Arow + kb + hi * 16);
    v8bf hh = *(const v8bf*)(Arow + kb + hi * 16 + 8);
    v16bf a = __builtin_shufflevector(lo, hh, 0, 1, 2, 3, 4, 5, 6, 7,
                                      8, 9, 10, 11, 12, 13, 14, 15);
    const __bf16* cp = &shc[l16 * SLDS + kb + hi * 16];
    v16bf c0 = *(const v16bf*)(cp);
    v16bf c1 = *(const v16bf*)(cp + 16 * SLDS);
    v16bf c2 = *(const v16bf*)(cp + 32 * SLDS);
    v16bf c3 = *(const v16bf*)(cp + 48 * SLDS);

    acc0 = __builtin_amdgcn_wmma_f32_16x16x32_bf16(false, a, false, c0, (short)0, acc0, false, false);
    acc1 = __builtin_amdgcn_wmma_f32_16x16x32_bf16(false, a, false, c1, (short)0, acc1, false, false);
    acc2 = __builtin_amdgcn_wmma_f32_16x16x32_bf16(false, a, false, c2, (short)0, acc2, false, false);
    acc3 = __builtin_amdgcn_wmma_f32_16x16x32_bf16(false, a, false, c3, (short)0, acc3, false, false);
  }

  // Epilogue: y = acc + D[d] * u[m,d]
#pragma unroll
  for (int t = 0; t < 4; t++) {
    const int d = dbase + t * 16 + l16;
    const float dd = Dv[d];
    const v8f acc = (t == 0) ? acc0 : (t == 1) ? acc1 : (t == 2) ? acc2 : acc3;
#pragma unroll
    for (int r = 0; r < 8; r++) {
      const int row = rbase + r + hi * 8;
      y[(size_t)row * DM + d] = acc[r] + dd * u[(size_t)row * DM + d];
    }
  }
}

// ---------------------------------------------------------------------------
extern "C" void kernel_launch(void* const* d_in, const int* in_sizes, int n_in,
                              void* d_out, int out_size, void* d_ws, size_t ws_size,
                              hipStream_t stream) {
  const float* u      = (const float*)d_in[0];
  const float* Lambda = (const float*)d_in[1];
  const float* log_dt = (const float*)d_in[2];
  const float* B      = (const float*)d_in[3];
  const float* C      = (const float*)d_in[4];
  const float* Dv     = (const float*)d_in[5];
  float* y = (float*)d_out;

  // workspace layout
  char* ws = (char*)d_ws;
  __bf16* Bb  = (__bf16*)(ws);                               // 128 KB
  __bf16* Cb  = (__bf16*)(ws + (1u << 17));                  // 128 KB
  float*  Bu  = (float*)(ws + (1u << 18));                   // 4 MB
  float*  S   = (float*)(ws + (1u << 18) + (4u << 20));      // 64 KB
  float*  Cin = (float*)(ws + (1u << 18) + (4u << 20) + (1u << 16));
  __bf16* Xb  = (__bf16*)(ws + (1u << 18) + (4u << 20) + (2u << 16)); // 2 MB

  // 1) weights -> bf16
  convert_weights<<<(2 * NS * DM) / 256, 256, 0, stream>>>(B, C, Bb, Cb);
  // 2) Bu = u @ B^T   (bf16 WMMA, LDS-staged B)
  gemm1_bu<<<M / 128, 128, 0, stream>>>(u, Bb, Bu);
  // 3) chunked linear-recurrence scan
  scan_local<<<BATCH * NCHUNK, CHUNK, 0, stream>>>(Bu, S, Lambda, log_dt);
  scan_carries<<<BATCH, NS, 0, stream>>>(S, Cin, Lambda, log_dt);
  fixup_bf16<<<(M * NS) / 256, 256, 0, stream>>>(Bu, Cin, Lambda, log_dt, Xb);
  // 4) y = xs @ C^T + D*u   (bf16 WMMA, LDS-staged C)
  gemm2_y<<<dim3(M / 128, DM / 64), 256, 0, stream>>>(Xb, Cb, u, Dv, y);
}
